// node_encode_attention_15015205666901
// MI455X (gfx1250) — matrix operand order repaired and checked by reading, not compiled, not run on hardware
//
#include <hip/hip_runtime.h>
#include <hip/hip_bf16.h>

#define B_    8
#define T_    64
#define N_    1024
#define F_    96
#define D_    32
#define H_    64
#define INIT_ 4
#define TE_   60   // effective time steps (t = 4..63)

typedef _Float16 half8 __attribute__((ext_vector_type(8)));
typedef _Float16 v16h  __attribute__((ext_vector_type(16)));
typedef float    v8f   __attribute__((ext_vector_type(8)));

#define V8F_ZERO {0.f,0.f,0.f,0.f,0.f,0.f,0.f,0.f}

static __device__ __forceinline__ v16h cat16(half8 lo, half8 hi) {
  return __builtin_shufflevector(lo, hi, 0,1,2,3,4,5,6,7,8,9,10,11,12,13,14,15);
}

// A-fragment (16x32 f16, MxK): lane<16 -> row=lane, K {0..7,16..23};
// lane>=16 -> row=lane-15... row=lane&15, K {8..15,24..31}. Contiguous b128 pairs.
static __device__ __forceinline__ v16h load_a_frag(const _Float16* base, int pitch) {
  int l = threadIdx.x & 31;
  const _Float16* p = base + (size_t)(l & 15) * pitch + (l >> 4) * 8;
  half8 lo = *(const half8*)p;
  half8 hi = *(const half8*)(p + 16);
  return cat16(lo, hi);
}

// B-fragment (32x16 f16, KxN): lane -> N col=lane&15, K {0..7,16..23} (lane<16) / {8..15,24..31}.
static __device__ __forceinline__ v16h load_b_frag(const _Float16* base, int pitch) {
  int l = threadIdx.x & 31;
  const _Float16* p = base + (l & 15) + (size_t)((l >> 4) * 8) * pitch;
  v16h v;
#pragma unroll
  for (int s = 0; s < 8; s++) {
    v[s] = p[(size_t)s * pitch];
    v[8 + s] = p[(size_t)(16 + s) * pitch];
  }
  return v;
}

static __device__ __forceinline__ v8f wmma32(v16h a, v16h b, v8f c) {
  return __builtin_amdgcn_wmma_f32_16x16x32_f16(false, a, false, b, (short)0, c, false, false);
}

// ---------- degree sums (rows of A_hat for lap_f, cols for lap_b) ----------
__global__ __launch_bounds__(256) void k_rowdeg(const float* __restrict__ adj,
                                               float* __restrict__ degr) {
  int t = INIT_ + blockIdx.y;
  int wid = threadIdx.x >> 5, lane = threadIdx.x & 31;
  int m = blockIdx.x * 8 + wid;
  const float* row = adj + ((size_t)t * N_ + m) * N_;
  float s = 0.f;
  for (int i = lane; i < N_; i += 32) s += row[i];
  for (int o = 16; o > 0; o >>= 1) s += __shfl_xor(s, o, 32);
  if (lane == 0) degr[t * N_ + m] = s + 1.0f;  // + self loop
}

__global__ __launch_bounds__(256) void k_coldeg(const float* __restrict__ adj,
                                               float* __restrict__ degc) {
  int t = INIT_ + blockIdx.y;
  int n = blockIdx.x * 256 + threadIdx.x;
  const float* base = adj + (size_t)t * N_ * N_ + n;
  float s = 0.f;
  for (int m = 0; m < N_; m++) s += base[(size_t)m * N_];
  degc[t * N_ + n] = s + 1.0f;
}

// ---------- build lap_f (row-major) and lap_b (transposed scatter via LDS) ----------
__global__ __launch_bounds__(256) void k_lap(const float* __restrict__ adj,
                                             const float* __restrict__ degr,
                                             const float* __restrict__ degc,
                                             _Float16* __restrict__ lapf,
                                             _Float16* __restrict__ lapb) {
  __shared__ __attribute__((aligned(16))) _Float16 tr[64][72];
  int t = INIT_ + blockIdx.z;
  int m0 = blockIdx.y * 64, n0 = blockIdx.x * 64;
  const float* ab = adj + (size_t)t * N_ * N_;
  const float* dr = degr + t * N_;
  const float* dc = degc + t * N_;
#pragma unroll 4
  for (int i = 0; i < 16; i++) {
    int flat = threadIdx.x + 256 * i;
    int r = flat >> 6, c = flat & 63;
    int m = m0 + r, n = n0 + c;
    float a  = ab[(size_t)m * N_ + n];
    float ah = a + ((m == n) ? 1.0f : 0.0f);
    float lf = rsqrtf(fmaxf(dr[m], 1e-6f)) * ah * rsqrtf(fmaxf(dr[n], 1e-6f));
    lapf[(size_t)t * N_ * N_ + (size_t)m * N_ + n] = (_Float16)lf;
    // lap_b[n,m] = dinvb[n] * (adj[m,n] + delta) * dinvb[m]
    float lb = rsqrtf(fmaxf(dc[n], 1e-6f)) * ah * rsqrtf(fmaxf(dc[m], 1e-6f));
    tr[c][r] = (_Float16)lb;
  }
  __syncthreads();
#pragma unroll 4
  for (int i = 0; i < 16; i++) {
    int flat = threadIdx.x + 256 * i;
    int r = flat >> 6, c = flat & 63;
    lapb[(size_t)t * N_ * N_ + (size_t)(n0 + r) * N_ + (m0 + c)] = tr[r][c];
  }
}

// ---------- fold weights: W_fm=W_f@Wm_top, W_bm=W_b@Wm_bot, transposed GRU weights ----------
__global__ __launch_bounds__(256) void k_prep(
    const float* Wf, const float* Wb, const float* Wm,
    const float* bf, const float* bb, const float* bm,
    const float* Wih, const float* Whh, const float* bih, const float* bhh,
    const float* Wout,
    _Float16* Wfm, _Float16* Wbm, float* bvec,
    _Float16* WihT, _Float16* WhhT, float* bgsum, _Float16* Wout16) {
  int tid = threadIdx.x;
  for (int idx = tid; idx < D_ * H_; idx += 256) {
    int i = idx >> 6, j = idx & 63;
    float s1 = 0.f, s2 = 0.f;
    for (int k = 0; k < H_; k++) {
      s1 += Wf[i * H_ + k] * Wm[k * H_ + j];
      s2 += Wb[i * H_ + k] * Wm[(H_ + k) * H_ + j];
    }
    Wfm[idx] = (_Float16)s1;
    Wbm[idx] = (_Float16)s2;
  }
  for (int j = tid; j < H_; j += 256) {
    float s = bm[j];
    for (int k = 0; k < H_; k++) s += bf[k] * Wm[k * H_ + j] + bb[k] * Wm[(H_ + k) * H_ + j];
    bvec[j] = s;
  }
  for (int idx = tid; idx < D_ * 3 * H_; idx += 256) {
    int k = idx / (3 * H_), g = idx % (3 * H_);
    WihT[idx] = (_Float16)Wih[g * D_ + k];
  }
  for (int idx = tid; idx < H_ * 3 * H_; idx += 256) {
    int k = idx / (3 * H_), g = idx % (3 * H_);
    WhhT[idx] = (_Float16)Whh[g * H_ + k];
  }
  for (int j = tid; j < 3 * H_; j += 256) bgsum[j] = bih[j] + bhh[j];
  for (int idx = tid; idx < H_ * D_; idx += 256) Wout16[idx] = (_Float16)Wout[idx];
}

// ---------- xe = x[...,32:96] @ W_enc  (f16 out; reassociated encoder GCN) ----------
__global__ __launch_bounds__(256) void k_xe(const float* __restrict__ x,
                                            const float* __restrict__ Wenc,
                                            _Float16* __restrict__ xe) {
  __shared__ float sW[64 * 32];
  __shared__ float sx[8][64];
  for (int i = threadIdx.x; i < 64 * 32; i += 256) sW[i] = Wenc[i];
  int b = blockIdx.y / TE_;
  int t = INIT_ + blockIdx.y % TE_;
  int n0 = blockIdx.x * 8;
  int d = threadIdx.x & 31, rr = threadIdx.x >> 5;
  const float* xr = x + ((size_t)(b * T_ + t) * N_ + n0 + rr) * F_ + D_;
  sx[rr][d] = xr[d];
  sx[rr][32 + d] = xr[32 + d];
  __syncthreads();
  float acc = 0.f;
#pragma unroll
  for (int f = 0; f < 64; f++) acc += sx[rr][f] * sW[f * 32 + d];
  xe[((size_t)(b * T_ + t) * N_ + n0 + rr) * D_ + d] = (_Float16)acc;
}

// ---------- ne = lap_f @ xe + b_enc (WMMA, one wave = one 16x32 output tile) ----------
__global__ __launch_bounds__(32) void k_ne(const _Float16* __restrict__ lapf,
                                           const _Float16* __restrict__ xe,
                                           const float* __restrict__ benc,
                                           float* __restrict__ ne) {
  const int lane = threadIdx.x & 31;
  const int b = blockIdx.x & 7;                 // b fastest -> L2 reuse of lap slice
  const int m0 = (blockIdx.x >> 3) * 16;
  const int t = INIT_ + blockIdx.y;
  const _Float16* Abase = lapf + ((size_t)t * N_ + m0) * N_;
  const _Float16* Bbase = xe + (size_t)(b * T_ + t) * N_ * D_;
  v8f a0 = V8F_ZERO, a1 = V8F_ZERO;
  for (int k = 0; k < N_; k += 32) {
    v16h a  = load_a_frag(Abase + k, N_);
    v16h b0 = load_b_frag(Bbase + (size_t)k * D_, D_);
    v16h b1 = load_b_frag(Bbase + (size_t)k * D_ + 16, D_);
    __builtin_prefetch(Abase + k + 32, 0, 1);
    a0 = wmma32(a, b0, a0);
    a1 = wmma32(a, b1, a1);
  }
  int col = lane & 15, mo = (lane >> 4) * 8;
  float b0v = benc[col], b1v = benc[16 + col];
  float* ob = ne + (size_t)(b * T_ + t) * N_ * D_ + (size_t)m0 * D_;
#pragma unroll
  for (int r = 0; r < 8; r++) {
    ob[(size_t)(mo + r) * D_ + col]      = a0[r] + b0v;
    ob[(size_t)(mo + r) * D_ + 16 + col] = a1[r] + b1v;
  }
}

// ---------- softmax over node dim (axis=2) + tmp = alpha * x[...,:32] (f16) ----------
__global__ __launch_bounds__(256) void k_soft(const float* __restrict__ ne,
                                              const float* __restrict__ x,
                                              _Float16* __restrict__ tmp16) {
  __shared__ float sm[8][32];
  __shared__ float ss[8][32];
  int b = blockIdx.x / TE_;
  int t = INIT_ + blockIdx.x % TE_;
  int d = threadIdx.x & 31, g = threadIdx.x >> 5;
  const float* nb = ne + (size_t)(b * T_ + t) * N_ * D_;
  float mx = -3.402823e38f;
  for (int n = g; n < N_; n += 8) mx = fmaxf(mx, nb[(size_t)n * D_ + d]);
  sm[g][d] = mx;
  __syncthreads();
  if (g == 0) {
    float m2 = sm[0][d];
    for (int j = 1; j < 8; j++) m2 = fmaxf(m2, sm[j][d]);
    sm[0][d] = m2;
  }
  __syncthreads();
  mx = sm[0][d];
  float s = 0.f;
  for (int n = g; n < N_; n += 8) s += __expf(nb[(size_t)n * D_ + d] - mx);
  ss[g][d] = s;
  __syncthreads();
  if (g == 0) {
    float s2 = 0.f;
    for (int j = 0; j < 8; j++) s2 += ss[j][d];
    ss[0][d] = s2;
  }
  __syncthreads();
  float rinv = 1.f / ss[0][d];
  const float* xb = x + (size_t)(b * T_ + t) * N_ * F_;
  _Float16* ob = tmp16 + (size_t)(b * T_ + t) * N_ * D_;
  for (int n = g; n < N_; n += 8) {
    float a = __expf(nb[(size_t)n * D_ + d] - mx) * rinv;
    ob[(size_t)n * D_ + d] = (_Float16)(a * xb[(size_t)n * F_ + d]);
  }
}

// ---------- mega kernel: u_f/u_b GCNs + h_space + GRU cell + output projection ----------
__global__ __launch_bounds__(32) void k_fused(
    const _Float16* __restrict__ lapf, const _Float16* __restrict__ lapb,
    const _Float16* __restrict__ tmp16,
    const _Float16* __restrict__ Wfm, const _Float16* __restrict__ Wbm,
    const float* __restrict__ bvec,
    const _Float16* __restrict__ WihT, const _Float16* __restrict__ WhhT,
    const float* __restrict__ bgsum, const float* __restrict__ bih,
    const float* __restrict__ bhh, const _Float16* __restrict__ Wout16,
    const float* __restrict__ bout, float* __restrict__ out) {
  // pitch 72 halves (36 dwords): row starts hit distinct banks for 16 rows
  __shared__ __attribute__((aligned(16))) _Float16 sU[2 * 16 * 72];
  __shared__ __attribute__((aligned(16))) _Float16 sH[16 * 72];
  const int lane = threadIdx.x & 31;
  const int b = blockIdx.x & 7;                 // b fastest -> lap slice reused in L2
  const int m0 = (blockIdx.x >> 3) * 16;
  const int t = INIT_ + blockIdx.y;

  const _Float16* Af = lapf + ((size_t)t * N_ + m0) * N_;
  const _Float16* Ab = lapb + ((size_t)t * N_ + m0) * N_;
  const _Float16* Tb = tmp16 + (size_t)(b * T_ + t) * N_ * D_;

  // ---- u_f = lap_f @ tmp, u_b = lap_b @ tmp (K=1024, f32 acc) ----
  v8f uf0 = V8F_ZERO, uf1 = V8F_ZERO, ub0 = V8F_ZERO, ub1 = V8F_ZERO;
  for (int k = 0; k < N_; k += 32) {
    v16h af = load_a_frag(Af + k, N_);
    v16h abf = load_a_frag(Ab + k, N_);
    v16h b0 = load_b_frag(Tb + (size_t)k * D_, D_);
    v16h b1 = load_b_frag(Tb + (size_t)k * D_ + 16, D_);
    __builtin_prefetch(Af + k + 32, 0, 1);
    __builtin_prefetch(Ab + k + 32, 0, 1);
    uf0 = wmma32(af, b0, uf0);
    uf1 = wmma32(af, b1, uf1);
    ub0 = wmma32(abf, b0, ub0);
    ub1 = wmma32(abf, b1, ub1);
  }

  const int col = lane & 15;
  const int mo = (lane >> 4) * 8;

  // ---- re-stage u tiles (D layout) as f16 A-fragments through LDS ----
#pragma unroll
  for (int r = 0; r < 8; r++) {
    int row = mo + r;
    sU[row * 72 + col]                 = (_Float16)uf0[r];
    sU[row * 72 + 16 + col]            = (_Float16)uf1[r];
    sU[16 * 72 + row * 72 + col]       = (_Float16)ub0[r];
    sU[16 * 72 + row * 72 + 16 + col]  = (_Float16)ub1[r];
  }
  asm volatile("s_wait_dscnt 0" ::: "memory");
  __syncthreads();
  v16h auf = load_a_frag(sU, 72);
  v16h aub = load_a_frag(sU + 16 * 72, 72);

  // ---- h_space = u_f@W_fm + u_b@W_bm + bvec (16x64) ----
  v8f hs[4];
#pragma unroll
  for (int j = 0; j < 4; j++) {
    v8f c = V8F_ZERO;
    c = wmma32(auf, load_b_frag(Wfm + j * 16, H_), c);
    c = wmma32(aub, load_b_frag(Wbm + j * 16, H_), c);
    float bv = bvec[j * 16 + col];
#pragma unroll
    for (int r = 0; r < 8; r++) c[r] += bv;
    hs[j] = c;
  }

  // ---- stage h_space for use as GEMM A operand ----
#pragma unroll
  for (int j = 0; j < 4; j++)
#pragma unroll
    for (int r = 0; r < 8; r++) sH[(mo + r) * 72 + j * 16 + col] = (_Float16)hs[j][r];
  asm volatile("s_wait_dscnt 0" ::: "memory");
  __syncthreads();
  v16h ah0 = load_a_frag(sH, 72);        // K = 0..31
  v16h ah1 = load_a_frag(sH + 32, 72);   // K = 32..63
  v16h atm = load_a_frag(Tb + (size_t)m0 * D_, D_);  // tmp tile (A operand for gi)
  __syncthreads();

  // ---- GRU gates; hidden written back into sH ----
#pragma unroll
  for (int j = 0; j < 4; j++) {
    int gr = j * 16, gz = H_ + j * 16, gn = 2 * H_ + j * 16;
    v8f cr = V8F_ZERO, cz = V8F_ZERO, ci = V8F_ZERO, ch = V8F_ZERO;
    cr = wmma32(atm, load_b_frag(WihT + gr, 3 * H_), cr);
    cr = wmma32(ah0, load_b_frag(WhhT + gr, 3 * H_), cr);
    cr = wmma32(ah1, load_b_frag(WhhT + 32 * 3 * H_ + gr, 3 * H_), cr);
    cz = wmma32(atm, load_b_frag(WihT + gz, 3 * H_), cz);
    cz = wmma32(ah0, load_b_frag(WhhT + gz, 3 * H_), cz);
    cz = wmma32(ah1, load_b_frag(WhhT + 32 * 3 * H_ + gz, 3 * H_), cz);
    ci = wmma32(atm, load_b_frag(WihT + gn, 3 * H_), ci);
    ch = wmma32(ah0, load_b_frag(WhhT + gn, 3 * H_), ch);
    ch = wmma32(ah1, load_b_frag(WhhT + 32 * 3 * H_ + gn, 3 * H_), ch);
    float bgr = bgsum[gr + col], bgz = bgsum[gz + col];
    float bin = bih[gn + col], bhn = bhh[gn + col];
#pragma unroll
    for (int r = 0; r < 8; r++) {
      float rg = 1.f / (1.f + __expf(-(cr[r] + bgr)));
      float zg = 1.f / (1.f + __expf(-(cz[r] + bgz)));
      float nn = tanhf((ci[r] + bin) + rg * (ch[r] + bhn));
      float h = (1.f - zg) * nn + zg * hs[j][r];
      sH[(mo + r) * 72 + j * 16 + col] = (_Float16)h;
    }
  }
  asm volatile("s_wait_dscnt 0" ::: "memory");
  __syncthreads();

  // ---- out = hidden @ W_out + b_out ----
  v16h ahd0 = load_a_frag(sH, 72);
  v16h ahd1 = load_a_frag(sH + 32, 72);
  v8f o0 = V8F_ZERO, o1 = V8F_ZERO;
  o0 = wmma32(ahd0, load_b_frag(Wout16, D_), o0);
  o0 = wmma32(ahd1, load_b_frag(Wout16 + 32 * D_, D_), o0);
  o1 = wmma32(ahd0, load_b_frag(Wout16 + 16, D_), o1);
  o1 = wmma32(ahd1, load_b_frag(Wout16 + 32 * D_ + 16, D_), o1);
  float bo0 = bout[col], bo1 = bout[16 + col];
  float* ob = out + (((size_t)b * TE_ + (t - INIT_)) * N_ + m0) * D_;
#pragma unroll
  for (int r = 0; r < 8; r++) {
    ob[(size_t)(mo + r) * D_ + col]      = o0[r] + bo0;
    ob[(size_t)(mo + r) * D_ + 16 + col] = o1[r] + bo1;
  }
}

extern "C" void kernel_launch(void* const* d_in, const int* in_sizes, int n_in,
                              void* d_out, int out_size, void* d_ws, size_t ws_size,
                              hipStream_t stream) {
  const float* x    = (const float*)d_in[0];
  const float* adj  = (const float*)d_in[1];
  const float* Wenc = (const float*)d_in[2];
  const float* benc = (const float*)d_in[3];
  const float* Wf   = (const float*)d_in[4];
  const float* bf   = (const float*)d_in[5];
  const float* Wb   = (const float*)d_in[6];
  const float* bb   = (const float*)d_in[7];
  const float* Wm   = (const float*)d_in[8];
  const float* bm   = (const float*)d_in[9];
  const float* Wih  = (const float*)d_in[10];
  const float* Whh  = (const float*)d_in[11];
  const float* bih  = (const float*)d_in[12];
  const float* bhh  = (const float*)d_in[13];
  const float* Wout = (const float*)d_in[14];
  const float* bout = (const float*)d_in[15];
  float* out = (float*)d_out;

  char* w = (char*)d_ws;
  size_t off = 0;
  auto take = [&](size_t bytes) -> char* {
    char* r = w + off;
    off = (off + bytes + 255) & ~(size_t)255;
    return r;
  };
  _Float16* lapf  = (_Float16*)take((size_t)T_ * N_ * N_ * 2);   // 128 MB
  _Float16* lapb  = (_Float16*)take((size_t)T_ * N_ * N_ * 2);   // 128 MB
  _Float16* xe16  = (_Float16*)take((size_t)B_ * T_ * N_ * D_ * 2);
  _Float16* tmp16 = (_Float16*)take((size_t)B_ * T_ * N_ * D_ * 2);
  float*    ne32  = (float*)take((size_t)B_ * T_ * N_ * D_ * 4); // 64 MB
  float*    degr  = (float*)take((size_t)T_ * N_ * 4);
  float*    degc  = (float*)take((size_t)T_ * N_ * 4);
  _Float16* Wfm16   = (_Float16*)take(D_ * H_ * 2);
  _Float16* Wbm16   = (_Float16*)take(D_ * H_ * 2);
  float*    bvec    = (float*)take(H_ * 4);
  _Float16* WihT16  = (_Float16*)take(D_ * 3 * H_ * 2);
  _Float16* WhhT16  = (_Float16*)take(H_ * 3 * H_ * 2);
  float*    bgsum   = (float*)take(3 * H_ * 4);
  _Float16* Wout16w = (_Float16*)take(H_ * D_ * 2);

  k_rowdeg<<<dim3(N_ / 8, TE_), 256, 0, stream>>>(adj, degr);
  k_coldeg<<<dim3(N_ / 256, TE_), 256, 0, stream>>>(adj, degc);
  k_lap<<<dim3(N_ / 64, N_ / 64, TE_), 256, 0, stream>>>(adj, degr, degc, lapf, lapb);
  k_prep<<<1, 256, 0, stream>>>(Wf, Wb, Wm, bf, bb, bm, Wih, Whh, bih, bhh, Wout,
                                Wfm16, Wbm16, bvec, WihT16, WhhT16, bgsum, Wout16w);
  k_xe<<<dim3(N_ / 8, B_ * TE_), 256, 0, stream>>>(x, Wenc, xe16);
  k_ne<<<dim3((N_ / 16) * B_, TE_), 32, 0, stream>>>(lapf, xe16, benc, ne32);
  k_soft<<<dim3(B_ * TE_), 256, 0, stream>>>(ne32, x, tmp16);
  k_fused<<<dim3((N_ / 16) * B_, TE_), 32, 0, stream>>>(
      lapf, lapb, tmp16, Wfm16, Wbm16, bvec, WihT16, WhhT16, bgsum, bih, bhh,
      Wout16w, bout, out);
}
